// Model_46548855554724
// MI455X (gfx1250) — matrix operand order repaired and checked by reading, not compiled, not run on hardware
//
#include <hip/hip_runtime.h>
#include <hip/hip_bf16.h>
#include <cstdint>
#include <cstddef>

// ---------------------------------------------------------------------------
// MI455X (gfx1250) fused MLP:
//   out  = relu(LN(relu(LN(X@W0+b0))@W1+b1))@W2+b2
//   coef = softmax(tanh(h_last@P)@cols)
// Mono-kernel: each block owns 32 rows; all activations live in LDS
// (X panel -> h1 -> h_last/h2). HBM traffic = X + out + weights (~440MB ->
// ~19us @ 23.3TB/s); compute ~0.44 TFLOP bf16 via v_wmma_f32_16x16x32_bf16.
// M=32 row-tile doubles L2 arithmetic intensity (32 FLOP/B): each packed
// B-fragment load feeds two WMMAs from registers, halving L2 weight traffic
// (12.8 GB total) vs an M=16 tiling.
// ---------------------------------------------------------------------------

typedef __attribute__((ext_vector_type(16))) __bf16 v16bf;
typedef __attribute__((ext_vector_type(8)))  float  v8f;

#define MROWS    32            // rows per block (2 WMMA row-tiles)
#define LDA      1032          // LDS panel row stride in bf16 (8-elem pad)
#define D_DIM    1024
#define H_DIM    1024
#define O_DIM    64
#define RANK_D   32
#define VIEWS_D  4
#define N_TOTAL  100000

// Dynamic LDS layout (bytes)
#define SM_PANEL_BYTES (MROWS * LDA * 2)                  // 66048
#define SM_PANEL0   0
#define SM_PANEL1   SM_PANEL_BYTES
#define SM_RED      (2 * SM_PANEL_BYTES)                  // redS/redQ/mean/rstd [32] each
#define SM_COEFT    (SM_RED + 512)                        // float[32][32]
#define SM_VIEWS    (SM_COEFT + 4096)                     // float[32][4]
#define SM_BYTES    (SM_VIEWS + 512)                      // 137216

// ---------------------------------------------------------------------------
// Pack a row-major f32 weight [K x Ncols] into per-lane WMMA-B bf16 fragments:
// frag(kt,nt): lane -> N = nt*16 + (lane&15), K = (lane>>4)*8 + {0..7, 16..23},
// 16 contiguous bf16 per lane -> B operand loads become two coalesced b128s.
// ---------------------------------------------------------------------------
__global__ void pack_weight_kernel(const float* __restrict__ W,
                                   __bf16* __restrict__ Wp,
                                   int KT, int NT, int Ncols) {
  int id   = blockIdx.x * blockDim.x + threadIdx.x;
  int lane = id & 31;
  int tile = id >> 5;
  if (tile >= KT * NT) return;
  int kt = tile / NT, nt = tile % NT;
  int n  = nt * 16 + (lane & 15);
  int kb = kt * 32 + (lane >> 4) * 8;
  __bf16* dst = Wp + ((size_t)tile * 32 + lane) * 16;
#pragma unroll
  for (int j = 0; j < 16; ++j)
    dst[j] = (__bf16)W[(size_t)(kb + (j < 8 ? j : j + 8)) * Ncols + n];
}

__device__ __forceinline__ v8f wmma_bf16(v16bf a, v16bf b, v8f c) {
  return __builtin_amdgcn_wmma_f32_16x16x32_bf16(false, a, false, b,
                                                 (short)0, c, false, false);
}

// ---------------------------------------------------------------------------
// Dual row-tile panel GEMM: A = 32 x 1024 bf16 panel in LDS, B = packed tiles.
// A-fragment (ISA 7.12.2, 16-bit A 16x32): lane&15 = row M, lane>>4 selects
// K offset {0,8}; halves at K and K+16 -> two ds b128 loads per fragment.
// Each B fragment (two global b128s) is reused by both row-tiles.
// ---------------------------------------------------------------------------
template<int CT>
__device__ __forceinline__ void gemm_rowpanel2(const __bf16* pA,
                                               const __bf16* __restrict__ Wp,
                                               int NT, int nt0, int lane,
                                               v8f* acc0, v8f* acc1) {
  const int row  = lane & 15;
  const int koff = (lane >> 4) * 8;
  const __bf16* arow0 = pA + row * LDA + koff;
  const __bf16* arow1 = pA + (row + 16) * LDA + koff;
#pragma unroll 4
  for (int kt = 0; kt < 32; ++kt) {
    v16bf a0, a1;
    const __bf16* ap0 = arow0 + kt * 32;
    const __bf16* ap1 = arow1 + kt * 32;
    ((uint4*)&a0)[0] = *(const uint4*)(ap0);
    ((uint4*)&a0)[1] = *(const uint4*)(ap0 + 16);
    ((uint4*)&a1)[0] = *(const uint4*)(ap1);
    ((uint4*)&a1)[1] = *(const uint4*)(ap1 + 16);
#pragma unroll
    for (int ct = 0; ct < CT; ++ct) {
      const __bf16* bp = Wp + (((size_t)kt * NT + nt0 + ct) * 32 + lane) * 16;
      v16bf b;
      ((uint4*)&b)[0] = *(const uint4*)(bp);
      ((uint4*)&b)[1] = *(const uint4*)(bp + 8);
      acc0[ct] = wmma_bf16(a0, b, acc0[ct]);
      acc1[ct] = wmma_bf16(a1, b, acc1[ct]);
    }
  }
}

// ---------------------------------------------------------------------------
// Fused bias + LayerNorm + ReLU epilogue over a 32 x 1024 block-wide result.
// C/D layout: lane 0-15 -> M=v, lane 16-31 -> M=v+8; N = lane&15 within tile.
// Row stats: shfl_xor tree within 16-lane halves, then 2 lanes/wave ds-atomic
// into a 32-entry LDS array, block barrier, apply.
// ---------------------------------------------------------------------------
__device__ __forceinline__ void ln_epilogue(
    v8f* acc0, v8f* acc1,
    const float* __restrict__ bias, const float* __restrict__ gamma,
    const float* __restrict__ beta,
    __bf16* outPanel, __bf16* prePanel,
    float* redS, float* redQ, float* meanA, float* rstdA,
    int wave, int lane, int tid) {
  const int colb  = wave * 128 + (lane & 15);
  const int rhalf = (lane >> 4) * 8;

#pragma unroll
  for (int ct = 0; ct < 8; ++ct) {
    const int col = colb + ct * 16;
    const float b = bias[col];
#pragma unroll
    for (int v = 0; v < 8; ++v) { acc0[ct][v] += b; acc1[ct][v] += b; }
    if (prePanel) {           // spill pre-norm h_last (bf16) for the P path
#pragma unroll
      for (int v = 0; v < 8; ++v) {
        prePanel[(rhalf + v) * LDA + col]      = (__bf16)acc0[ct][v];
        prePanel[(16 + rhalf + v) * LDA + col] = (__bf16)acc1[ct][v];
      }
    }
  }

  // ---- stats, tile0 then tile1 (sequential to limit live registers) ----
#pragma unroll
  for (int half = 0; half < 2; ++half) {
    v8f* acc = half ? acc1 : acc0;
    float s[8], q[8];
#pragma unroll
    for (int v = 0; v < 8; ++v) { s[v] = 0.f; q[v] = 0.f; }
#pragma unroll
    for (int ct = 0; ct < 8; ++ct)
#pragma unroll
      for (int v = 0; v < 8; ++v) { float f = acc[ct][v]; s[v] += f; q[v] += f * f; }
#pragma unroll
    for (int off = 1; off < 16; off <<= 1) {
#pragma unroll
      for (int v = 0; v < 8; ++v) {
        s[v] += __shfl_xor(s[v], off);
        q[v] += __shfl_xor(q[v], off);
      }
    }
    if ((lane & 15) == 0) {
#pragma unroll
      for (int v = 0; v < 8; ++v) {
        atomicAdd(&redS[half * 16 + rhalf + v], s[v]);
        atomicAdd(&redQ[half * 16 + rhalf + v], q[v]);
      }
    }
  }
  __syncthreads();
  if (tid < 32) {
    const float mu  = redS[tid] * (1.0f / (float)H_DIM);
    const float var = redQ[tid] * (1.0f / (float)H_DIM) - mu * mu;
    meanA[tid] = mu;
    rstdA[tid] = rsqrtf(var + 1e-5f);
    redS[tid] = 0.f; redQ[tid] = 0.f;   // re-arm for next LN
  }
  __syncthreads();
#pragma unroll
  for (int ct = 0; ct < 8; ++ct) {
    const int col = colb + ct * 16;
    const float g = gamma[col], bt = beta[col];
#pragma unroll
    for (int v = 0; v < 8; ++v) {
      const int r0 = rhalf + v, r1 = 16 + rhalf + v;
      float x0 = (acc0[ct][v] - meanA[r0]) * rstdA[r0] * g + bt;
      float x1 = (acc1[ct][v] - meanA[r1]) * rstdA[r1] * g + bt;
      x0 = x0 > 0.f ? x0 : 0.f;
      x1 = x1 > 0.f ? x1 : 0.f;
      outPanel[r0 * LDA + col] = (__bf16)x0;
      outPanel[r1 * LDA + col] = (__bf16)x1;
    }
  }
  __syncthreads();
}

__global__ __launch_bounds__(256, 1) void fused_mlp_kernel(
    const float* __restrict__ X,
    const float* __restrict__ b0, const float* __restrict__ g0, const float* __restrict__ bt0,
    const float* __restrict__ b1, const float* __restrict__ g1, const float* __restrict__ bt1,
    const float* __restrict__ b2, const float* __restrict__ colsW,
    const __bf16* __restrict__ W0p, const __bf16* __restrict__ W1p,
    const __bf16* __restrict__ W2p, const __bf16* __restrict__ Pp,
    float* __restrict__ out) {
  extern __shared__ __align__(16) char smem[];
  __bf16* panel0 = (__bf16*)(smem + SM_PANEL0);   // X  -> later h_last
  __bf16* panel1 = (__bf16*)(smem + SM_PANEL1);   // h1 -> later h2
  float* redS   = (float*)(smem + SM_RED);
  float* redQ   = redS + 32;
  float* meanA  = redQ + 32;
  float* rstdA  = meanA + 32;
  float* coefT  = (float*)(smem + SM_COEFT);      // tanh(h_last@P) [32][32]
  float* viewsS = (float*)(smem + SM_VIEWS);      // [32][4]

  const int tid  = threadIdx.x;
  const int lane = tid & 31;
  const int wave = tid >> 5;
  const int rowbase = blockIdx.x * MROWS;
  const int rhalf = (lane >> 4) * 8;

  if (tid < 32) { redS[tid] = 0.f; redQ[tid] = 0.f; }

  // Stage X row-panel f32 -> bf16 into LDS (coalesced float4 loads)
#pragma unroll
  for (int i = 0; i < 32; ++i) {
    const int idx = tid + i * 256;          // 8192 float4s = 32x1024 f32
    const int r   = idx >> 8;
    const int c4  = (idx & 255) * 4;
    const float4 v = *(const float4*)(X + (size_t)(rowbase + r) * D_DIM + c4);
    __bf16* d = panel0 + r * LDA + c4;
    d[0] = (__bf16)v.x; d[1] = (__bf16)v.y; d[2] = (__bf16)v.z; d[3] = (__bf16)v.w;
  }
  __syncthreads();

  v8f acc0[8], acc1[8];
  const v8f vzero = {0.f, 0.f, 0.f, 0.f, 0.f, 0.f, 0.f, 0.f};
#pragma unroll
  for (int ct = 0; ct < 8; ++ct) { acc0[ct] = vzero; acc1[ct] = vzero; }

  // ---- Layer 0: X @ W0 (+b0, LN, ReLU) -> h1 in panel1 ----
  gemm_rowpanel2<8>(panel0, W0p, 64, wave * 8, lane, acc0, acc1);
  ln_epilogue(acc0, acc1, b0, g0, bt0, panel1, nullptr,
              redS, redQ, meanA, rstdA, wave, lane, tid);

#pragma unroll
  for (int ct = 0; ct < 8; ++ct) { acc0[ct] = vzero; acc1[ct] = vzero; }

  // ---- Layer 1: h1 @ W1 ; h_last(pre-norm)->panel0, h2(LN+ReLU)->panel1 ----
  gemm_rowpanel2<8>(panel1, W1p, 64, wave * 8, lane, acc0, acc1);
  ln_epilogue(acc0, acc1, b1, g1, bt1, panel1, panel0,
              redS, redQ, meanA, rstdA, wave, lane, tid);

  // ---- Head: waves 0-3 -> out = h2@W2+b2 ; waves 4-5 -> tanh(h_last@P) ----
  if (wave < 4) {
    v8f c0 = vzero, c1 = vzero;
    gemm_rowpanel2<1>(panel1, W2p, 4, wave, lane, &c0, &c1);
    const int col = wave * 16 + (lane & 15);
    const float bb = b2[col];
#pragma unroll
    for (int v = 0; v < 8; ++v) {
      out[(size_t)(rowbase + rhalf + v) * O_DIM + col]      = c0[v] + bb;
      out[(size_t)(rowbase + 16 + rhalf + v) * O_DIM + col] = c1[v] + bb;
    }
  } else if (wave < 6) {
    v8f c0 = vzero, c1 = vzero;
    gemm_rowpanel2<1>(panel0, Pp, 2, wave - 4, lane, &c0, &c1);
    const int col = (wave - 4) * 16 + (lane & 15);
#pragma unroll
    for (int v = 0; v < 8; ++v) {
      coefT[(rhalf + v) * 32 + col]      = tanhf(c0[v]);
      coefT[(16 + rhalf + v) * 32 + col] = tanhf(c1[v]);
    }
  }
  __syncthreads();

  // coefs @ cols : 128 threads, one (row,view) each
  if (tid < 128) {
    const int r = tid >> 2, vw = tid & 3;
    float d = 0.f;
#pragma unroll
    for (int k = 0; k < RANK_D; ++k) d += coefT[r * 32 + k] * colsW[k * VIEWS_D + vw];
    viewsS[r * 4 + vw] = d;
  }
  __syncthreads();

  // row softmax over 4 views
  if (tid < 32) {
    const float v0 = viewsS[tid * 4 + 0], v1 = viewsS[tid * 4 + 1];
    const float v2 = viewsS[tid * 4 + 2], v3 = viewsS[tid * 4 + 3];
    const float m  = fmaxf(fmaxf(v0, v1), fmaxf(v2, v3));
    const float e0 = expf(v0 - m), e1 = expf(v1 - m);
    const float e2 = expf(v2 - m), e3 = expf(v3 - m);
    const float inv = 1.0f / (e0 + e1 + e2 + e3);
    float* o = out + (size_t)N_TOTAL * O_DIM + (size_t)(rowbase + tid) * VIEWS_D;
    o[0] = e0 * inv; o[1] = e1 * inv; o[2] = e2 * inv; o[3] = e3 * inv;
  }
}

// ---------------------------------------------------------------------------
extern "C" void kernel_launch(void* const* d_in, const int* in_sizes, int n_in,
                              void* d_out, int out_size, void* d_ws, size_t ws_size,
                              hipStream_t stream) {
  (void)in_sizes; (void)n_in; (void)out_size; (void)ws_size;
  const float* X    = (const float*)d_in[0];
  const float* W0   = (const float*)d_in[1];
  const float* b0   = (const float*)d_in[2];
  const float* g0   = (const float*)d_in[3];
  const float* bt0  = (const float*)d_in[4];
  const float* W1   = (const float*)d_in[5];
  const float* b1   = (const float*)d_in[6];
  const float* g1   = (const float*)d_in[7];
  const float* bt1  = (const float*)d_in[8];
  const float* W2   = (const float*)d_in[9];
  const float* b2   = (const float*)d_in[10];
  const float* P    = (const float*)d_in[11];
  const float* cols = (const float*)d_in[12];

  char* ws = (char*)d_ws;
  __bf16* W0p = (__bf16*)(ws);                              // 2 MB
  __bf16* W1p = (__bf16*)(ws + (size_t)2097152);            // 2 MB
  __bf16* W2p = (__bf16*)(ws + (size_t)4194304);            // 128 KB
  __bf16* Pp  = (__bf16*)(ws + (size_t)4325376);            // 64 KB

  // Pack weights into WMMA-B fragment layout (bf16). KT = K/32, NT = N/16.
  pack_weight_kernel<<<256, 256, 0, stream>>>(W0, W0p, 32, 64, H_DIM);
  pack_weight_kernel<<<256, 256, 0, stream>>>(W1, W1p, 32, 64, H_DIM);
  pack_weight_kernel<<<16,  256, 0, stream>>>(W2, W2p, 32, 4,  O_DIM);
  pack_weight_kernel<<<8,   256, 0, stream>>>(P,  Pp,  32, 2,  RANK_D);

  (void)hipFuncSetAttribute((const void*)fused_mlp_kernel,
                            hipFuncAttributeMaxDynamicSharedMemorySize,
                            SM_BYTES);
  fused_mlp_kernel<<<N_TOTAL / MROWS, 256, SM_BYTES, stream>>>(
      X, b0, g0, bt0, b1, g1, bt1, b2, cols,
      W0p, W1p, W2p, Pp, (float*)d_out);
}
